// HexWarGNN_88828513616304
// MI455X (gfx1250) — matrix operand order repaired and checked by reading, not compiled, not run on hardware
//
#include <hip/hip_runtime.h>
#include <hip/hip_bf16.h>
#include <stdint.h>

// ---------------------------------------------------------------------------
// HexWarGNN forward for MI455X (gfx1250, wave32, WMMA).
// Memory-bound on edge gathers => keep an fp16 mirror of h (halves gather
// bytes), run every dense GEMM through v_wmma_f32_16x16x32_f16, and stream
// the edge-MLP gather through GLOBAL_LOAD_ASYNC_TO_LDS_B128 (ASYNCcnt).
// ---------------------------------------------------------------------------

constexpr int kN      = 50000;
constexpr int kE      = 400000;
constexpr int kB      = 8;
constexpr int kHid    = 128;
constexpr int kHeads  = 4;
constexpr int kHdim   = 32;
constexpr int kLayers = 4;
constexpr int kNodeF  = 18;
constexpr int kGlobF  = 8;
constexpr int kEdgeF  = 6;
constexpr int kEdgeIn = 2 * kHid + kEdgeF;   // 262
constexpr int kKpad   = 288;                 // 262 padded to 9 K-tiles of 32

// input tensor order (setup_inputs dict insertion order, params flattened)
enum {
  IN_X = 0, IN_EIDX, IN_EATTR, IN_U, IN_MASK, IN_BATCH,
  IN_NE_W, IN_NE_B, IN_NE_G, IN_NE_BN,
  IN_GE_W, IN_GE_B, IN_GE_G, IN_GE_BN,
  IN_LAYER0 = 14,                      // per layer: Wl,bl,Wr,br,We,att,bias,g,bn
  IN_MOVE_W1 = IN_LAYER0 + 9 * kLayers, // 50
  IN_MOVE_B1, IN_MOVE_W2, IN_MOVE_B2,
  IN_FRAC_W1, IN_FRAC_B1, IN_FRAC_W2, IN_FRAC_B2,
  IN_VAL_W1, IN_VAL_B1, IN_VAL_W2, IN_VAL_B2
};

// -------------------------------- WMMA ------------------------------------
typedef _Float16 v16h __attribute__((ext_vector_type(16)));
typedef float    v8f  __attribute__((ext_vector_type(8)));

union Frag { v16h v; uint4 q[2]; };

__device__ __forceinline__ v8f wmma_f16(v16h a, v16h b, v8f c) {
  // emits v_wmma_f32_16x16x32_f16
  return __builtin_amdgcn_wmma_f32_16x16x32_f16(
      /*neg_a=*/false, a, /*neg_b=*/false, b,
      /*c_mod=*/(short)0, c, /*reuse_a=*/false, /*reuse_b=*/false);
}

// A-matrix 16x32 f16, row-major src with row stride ldk (elements).
// lane<16 : M=lane,    K = base+[0..7] and base+[16..23]
// lane>=16: M=lane-16, K = base+[8..15] and base+[24..31]
__device__ __forceinline__ v16h load_a_frag(const _Float16* a, int ldk, int kbase) {
  const int lane = threadIdx.x & 31;
  const int m  = lane & 15;
  const int ko = (lane < 16) ? 0 : 8;
  Frag f;
  const _Float16* r = a + (size_t)m * ldk + kbase + ko;
  f.q[0] = *(const uint4*)(r);
  f.q[1] = *(const uint4*)(r + 16);
  return f.v;
}

// B-matrix 32x16 f16 from a pre-transposed weight Wt[outCol][K] (row stride ldk).
// lane holds N = lane%16, K = base+[0..7],[16..23] (lane<16) or base+[8..15],[24..31].
__device__ __forceinline__ v16h load_b_frag(const _Float16* wt, int ldk,
                                            int nbase, int kbase) {
  const int lane = threadIdx.x & 31;
  const int n  = lane & 15;
  const int ko = (lane < 16) ? 0 : 8;
  Frag f;
  const _Float16* r = wt + (size_t)(nbase + n) * ldk + kbase + ko;
  f.q[0] = *(const uint4*)(r);
  f.q[1] = *(const uint4*)(r + 16);
  return f.v;
}

// -------------------------- async global->LDS path -------------------------
// Builtin parameter types (from the round-2 diagnostic): param 1 is
// 'int __vector(4) __device__ *' (AS1), param 2 the LDS (AS3) counterpart.
#if __has_builtin(__builtin_amdgcn_global_load_async_to_lds_b128)
#define HW_ASYNC_LDS 1
typedef int v4i_async __attribute__((vector_size(16)));
#define ASYNC_G(p) ((__attribute__((address_space(1))) v4i_async*)(void*)(p))
#define ASYNC_L(p) ((__attribute__((address_space(3))) v4i_async*)(void*)(p))
#endif

__device__ __forceinline__ void wait_async0() {
#if __has_builtin(__builtin_amdgcn_s_wait_asynccnt)
  __builtin_amdgcn_s_wait_asynccnt(0);
#else
  asm volatile("s_wait_asynccnt 0" ::: "memory");
#endif
}

// ------------------------------ math helpers -------------------------------
__device__ __forceinline__ float gelu_f(float x) {
  return 0.5f * x * (1.0f + erff(x * 0.70710678118654752f));
}
__device__ __forceinline__ float softplus_f(float x) {
  return (x > 20.f) ? x : log1pf(expf(x));
}
// order-preserving float<->uint key (all-zero init == below -inf)
__device__ __forceinline__ unsigned fkey(float f) {
  unsigned u = __float_as_uint(f);
  return (u & 0x80000000u) ? ~u : (u | 0x80000000u);
}
__device__ __forceinline__ float funkey(unsigned k) {
  unsigned u = (k & 0x80000000u) ? (k & 0x7FFFFFFFu) : ~k;
  return __uint_as_float(u);
}
__device__ __forceinline__ float block_reduce_sum_128(float v, float* sred) {
  const int t = threadIdx.x;
  sred[t] = v; __syncthreads();
  #pragma unroll
  for (int s = 64; s > 0; s >>= 1) {
    if (t < s) sred[t] += sred[t + s];
    __syncthreads();
  }
  float r = sred[0]; __syncthreads();
  return r;
}

// --------------------------- prep: weight transpose ------------------------
// W[K][Nout] fp32 row-major -> Wt[Nout][Kpad] fp16, zero K padding.
__global__ void convtr_kernel(const float* __restrict__ W, _Float16* __restrict__ Wt,
                              int K, int Nout, int Kpad) {
  int i = blockIdx.x * blockDim.x + threadIdx.x;
  if (i >= Nout * Kpad) return;
  int n = i / Kpad, k = i - n * Kpad;
  Wt[i] = (k < K) ? (_Float16)W[(size_t)k * Nout + n] : (_Float16)0.f;
}

// ------------------------------- encoders ----------------------------------
__global__ __launch_bounds__(128) void glob_enc_kernel(
    const float* __restrict__ u, const float* __restrict__ W,
    const float* __restrict__ b, const float* __restrict__ g,
    const float* __restrict__ bn, float* __restrict__ u_emb) {
  __shared__ float sred[128];
  const int bi = blockIdx.x, c = threadIdx.x;
  float acc = b[c];
  #pragma unroll
  for (int k = 0; k < kGlobF; ++k) acc += u[bi * kGlobF + k] * W[k * kHid + c];
  float mu  = block_reduce_sum_128(acc, sred) * (1.f / kHid);
  float d   = acc - mu;
  float var = block_reduce_sum_128(d * d, sred) * (1.f / kHid);
  u_emb[bi * kHid + c] = gelu_f(d * rsqrtf(var + 1e-5f) * g[c] + bn[c]);
}

__global__ __launch_bounds__(128) void node_enc_kernel(
    const float* __restrict__ x, const float* __restrict__ W,
    const float* __restrict__ b, const float* __restrict__ g,
    const float* __restrict__ bn, const float* __restrict__ u_emb,
    const int* __restrict__ batch, float* __restrict__ h,
    _Float16* __restrict__ hbf) {
  __shared__ float sred[128];
  const int n = blockIdx.x, c = threadIdx.x;
  float acc = b[c];
  #pragma unroll
  for (int k = 0; k < kNodeF; ++k) acc += x[(size_t)n * kNodeF + k] * W[k * kHid + c];
  float mu  = block_reduce_sum_128(acc, sred) * (1.f / kHid);
  float d   = acc - mu;
  float var = block_reduce_sum_128(d * d, sred) * (1.f / kHid);
  float o   = gelu_f(d * rsqrtf(var + 1e-5f) * g[c] + bn[c]) + u_emb[batch[n] * kHid + c];
  h[(size_t)n * kHid + c]   = o;
  hbf[(size_t)n * kHid + c] = (_Float16)o;
}

// ----------------------- per-layer GEMM: xl/xr = h@W + b -------------------
// 16 output rows per block; wave w owns 16 cols; 4 K-tiles x 2 weights.
__global__ __launch_bounds__(256) void gemm_xlxr_kernel(
    const _Float16* __restrict__ hbf, const _Float16* __restrict__ wlt,
    const _Float16* __restrict__ wrt, const float* __restrict__ bl,
    const float* __restrict__ br, float* __restrict__ xl, float* __restrict__ xr) {
  const int wave = threadIdx.x >> 5;
  const int row0 = blockIdx.x * 16;
  const int n0   = wave * 16;
  const _Float16* a_base = hbf + (size_t)row0 * kHid;
  __builtin_prefetch(a_base + 16 * kHid, 0, 1);    // global_prefetch_b8
  v8f cl = {}; v8f cr = {};
  #pragma unroll
  for (int kt = 0; kt < 4; ++kt) {
    v16h a  = load_a_frag(a_base, kHid, kt * 32);
    v16h bL = load_b_frag(wlt, kHid, n0, kt * 32);
    cl = wmma_f16(a, bL, cl);
    v16h bR = load_b_frag(wrt, kHid, n0, kt * 32);
    cr = wmma_f16(a, bR, cr);
  }
  const int lane = threadIdx.x & 31;
  const int n = lane & 15, mo = (lane < 16) ? 0 : 8;
  const float bLn = bl[n0 + n], bRn = br[n0 + n];
  #pragma unroll
  for (int r = 0; r < 8; ++r) {
    size_t o = (size_t)(row0 + mo + r) * kHid + n0 + n;
    xl[o] = cl[r] + bLn;
    xr[o] = cr[r] + bRn;
  }
}

// --------------------------- GATv2 edge passes -----------------------------
__global__ __launch_bounds__(256) void gat_logits_kernel(
    const float* __restrict__ xl, const float* __restrict__ xr,
    const int* __restrict__ eidx, const float* __restrict__ eattr,
    const float* __restrict__ We, const float* __restrict__ att,
    float* __restrict__ logits, unsigned* __restrict__ segmax) {
  const int idx = blockIdx.x * blockDim.x + threadIdx.x;
  if (idx >= kE * kHeads) return;
  const int e = idx >> 2, hd = idx & 3;
  const int src = eidx[e], dst = eidx[kE + e];
  const float* xls = xl + (size_t)src * kHid + hd * kHdim;
  const float* xrd = xr + (size_t)dst * kHid + hd * kHdim;
  float ea[kEdgeF];
  #pragma unroll
  for (int k = 0; k < kEdgeF; ++k) ea[k] = eattr[(size_t)e * kEdgeF + k];
  float logit = 0.f;
  #pragma unroll 4
  for (int j = 0; j < kHdim; ++j) {
    float ee = 0.f;
    #pragma unroll
    for (int k = 0; k < kEdgeF; ++k) ee += ea[k] * We[k * kHid + hd * kHdim + j];
    float m = xls[j] + xrd[j] + ee;
    m = (m > 0.f) ? m : 0.2f * m;                       // leaky_relu(0.2)
    logit += m * att[hd * kHdim + j];
  }
  logits[idx] = logit;
  atomicMax(segmax + (size_t)dst * kHeads + hd, fkey(logit));
}

__global__ __launch_bounds__(256) void gat_expsum_kernel(
    const int* __restrict__ eidx, float* __restrict__ logits,
    const unsigned* __restrict__ segmax, float* __restrict__ den) {
  const int idx = blockIdx.x * blockDim.x + threadIdx.x;
  if (idx >= kE * kHeads) return;
  const int e = idx >> 2, hd = idx & 3;
  const int dst = eidx[kE + e];
  float ex = expf(logits[idx] - funkey(segmax[(size_t)dst * kHeads + hd]));
  logits[idx] = ex;                                     // reuse buffer
  atomicAdd(den + (size_t)dst * kHeads + hd, ex);
}

__global__ __launch_bounds__(256) void gat_scatter_kernel(
    const int* __restrict__ eidx, const float* __restrict__ ex,
    const float* __restrict__ den, const float* __restrict__ xl,
    float* __restrict__ gat) {
  const int idx = blockIdx.x * blockDim.x + threadIdx.x;
  if (idx >= kE * kHeads) return;
  const int e = idx >> 2, hd = idx & 3;
  const int src = eidx[e], dst = eidx[kE + e];
  const float alpha = ex[idx] / den[(size_t)dst * kHeads + hd];
  const float* xls = xl + (size_t)src * kHid + hd * kHdim;
  float* o = gat + (size_t)dst * kHid + hd * kHdim;
  #pragma unroll 4
  for (int j = 0; j < kHdim; ++j) atomicAdd(o + j, xls[j] * alpha);
}

__global__ __launch_bounds__(128) void node_update_kernel(
    float* __restrict__ h, _Float16* __restrict__ hbf,
    const float* __restrict__ gat, const float* __restrict__ bias,
    const float* __restrict__ g, const float* __restrict__ bn) {
  __shared__ float sred[128];
  const int n = blockIdx.x, c = threadIdx.x;
  const size_t o = (size_t)n * kHid + c;
  float v   = h[o] + gat[o] + bias[c];                  // residual + (out+bias)
  float mu  = block_reduce_sum_128(v, sred) * (1.f / kHid);
  float d   = v - mu;
  float var = block_reduce_sum_128(d * d, sred) * (1.f / kHid);
  float out = d * rsqrtf(var + 1e-5f) * g[c] + bn[c];
  h[o]   = out;
  hbf[o] = (_Float16)out;
}

// ------------------------------ edge MLP heads -----------------------------
// 64 edges per block; emb = [h16[src] | h16[dst] | attr | pad] staged in LDS
// via async global->LDS gather, both move/frac 288->128 GEMMs via WMMA,
// then 128->{1,2} dot reduction.
constexpr int kEmbS    = 296;                          // padded LDS row stride
constexpr int kHidS    = 132;                          // padded hidden stride
constexpr int kSmemLen = 2 * 64 * kHidS * 4;           // 67584 >= 64*296*2

__global__ __launch_bounds__(256) void edge_mlp_kernel(
    const _Float16* __restrict__ hbf, const int* __restrict__ eidx,
    const float* __restrict__ eattr,
    const _Float16* __restrict__ w1mt, const float* __restrict__ b1m,
    const float* __restrict__ w2m, const float* __restrict__ b2m,
    const _Float16* __restrict__ w1ft, const float* __restrict__ b1f,
    const float* __restrict__ w2f, const float* __restrict__ b2f,
    float* __restrict__ out_move, float* __restrict__ out_alpha,
    float* __restrict__ out_beta) {
  __shared__ __align__(16) unsigned char smem[kSmemLen];
  _Float16* emb  = (_Float16*)smem;                    // [64][296]
  float*    hidm = (float*)smem;                       // [64][132] (after reuse)
  float*    hidf = (float*)(smem + 64 * kHidS * 4);
  const int tid = threadIdx.x;
  const int e0  = blockIdx.x * 64;

  { // gather: 4 threads per edge, 128B (64 fp16) each, global -> LDS
    const int eloc = tid >> 2, t4 = tid & 3;
    const int e = e0 + eloc;
    const int src = eidx[e], dst = eidx[kE + e];
    const _Float16* hrow =
        hbf + (size_t)((t4 < 2) ? src : dst) * kHid + (t4 & 1) * 64;
    _Float16* drow = emb + eloc * kEmbS + ((t4 < 2) ? 0 : kHid) + (t4 & 1) * 64;
#ifdef HW_ASYNC_LDS
    #pragma unroll
    for (int i = 0; i < 8; ++i)   // global_load_async_to_lds_b128, ASYNCcnt
      __builtin_amdgcn_global_load_async_to_lds_b128(
          ASYNC_G((const char*)hrow + i * 16), ASYNC_L((char*)drow + i * 16),
          /*offset=*/0, /*cpol=*/0);
#else
    const uint4* s = (const uint4*)hrow;
    uint4* d = (uint4*)drow;
    #pragma unroll
    for (int i = 0; i < 8; ++i) d[i] = s[i];
#endif
    if (t4 == 0) {
      _Float16* er = emb + eloc * kEmbS + 2 * kHid;
      #pragma unroll
      for (int k = 0; k < kEdgeF; ++k) er[k] = (_Float16)eattr[(size_t)e * kEdgeF + k];
      #pragma unroll
      for (int k = kEdgeF; k < 32; ++k) er[k] = (_Float16)0.f;
    }
  }
#ifdef HW_ASYNC_LDS
  wait_async0();                                        // s_wait_asynccnt 0
#endif
  __syncthreads();

  // WMMA: wave w -> row tile (w&3), col half (w>>2); 9 K-tiles x 4 N-tiles x 2
  const int wave = tid >> 5;
  const int rt = wave & 3;
  const int ncol0 = (wave >> 2) * 64;
  v8f accm[4] = {{0}, {0}, {0}, {0}};
  v8f accf[4] = {{0}, {0}, {0}, {0}};
  const _Float16* arow = emb + rt * 16 * kEmbS;
  for (int kt = 0; kt < 9; ++kt) {
    v16h a = load_a_frag(arow, kEmbS, kt * 32);        // ds_load_b128 x2
    #pragma unroll
    for (int nt = 0; nt < 4; ++nt) {
      v16h bm = load_b_frag(w1mt, kKpad, ncol0 + nt * 16, kt * 32);
      accm[nt] = wmma_f16(a, bm, accm[nt]);
      v16h bfv = load_b_frag(w1ft, kKpad, ncol0 + nt * 16, kt * 32);
      accf[nt] = wmma_f16(a, bfv, accf[nt]);
    }
  }
  __syncthreads();                                      // done reading emb

  { // bias + GELU, park hidden tiles in LDS (reuses emb region)
    const int lane = tid & 31, n = lane & 15, mo = (lane < 16) ? 0 : 8;
    #pragma unroll
    for (int nt = 0; nt < 4; ++nt) {
      const int col = ncol0 + nt * 16 + n;
      const float bm = b1m[col], bff = b1f[col];
      #pragma unroll
      for (int r = 0; r < 8; ++r) {
        const int row = rt * 16 + mo + r;
        hidm[row * kHidS + col] = gelu_f(accm[nt][r] + bm);
        hidf[row * kHidS + col] = gelu_f(accf[nt][r] + bff);
      }
    }
  }
  __syncthreads();

  { // second GEMM (128 -> 1/2) as lane-group-of-4 dot reduction
    const int eloc = tid >> 2, t4 = tid & 3;
    const float* hm = hidm + eloc * kHidS + t4 * 32;
    const float* hf = hidf + eloc * kHidS + t4 * 32;
    float pm = 0.f, pa = 0.f, pb = 0.f;
    #pragma unroll
    for (int k = 0; k < 32; ++k) {
      const int kk = t4 * 32 + k;
      pm += hm[k] * w2m[kk];
      const float hv = hf[k];
      pa += hv * w2f[kk * 2 + 0];
      pb += hv * w2f[kk * 2 + 1];
    }
    pm += __shfl_down(pm, 2, 4); pm += __shfl_down(pm, 1, 4);
    pa += __shfl_down(pa, 2, 4); pa += __shfl_down(pa, 1, 4);
    pb += __shfl_down(pb, 2, 4); pb += __shfl_down(pb, 1, 4);
    if (t4 == 0) {
      const int e = e0 + eloc;
      out_move[e]  = pm + b2m[0];
      out_alpha[e] = softplus_f(pa + b2f[0]) + 1e-4f;
      out_beta[e]  = softplus_f(pb + b2f[1]) + 1e-4f;
    }
  }
}

// ------------------------------- pooling + value ---------------------------
__global__ __launch_bounds__(256) void pool_kernel(
    const float* __restrict__ h, const int* __restrict__ batch,
    const unsigned char* __restrict__ mask, float* __restrict__ gsum,
    float* __restrict__ asum, float* __restrict__ cnt, float* __restrict__ acnt) {
  const int n = blockIdx.x * blockDim.x + threadIdx.x;
  if (n >= kN) return;
  const int b = batch[n];
  const float am = mask[n] ? 1.f : 0.f;
  atomicAdd(cnt + b, 1.f);
  atomicAdd(acnt + b, am);
  const float* hr = h + (size_t)n * kHid;
  for (int c = 0; c < kHid; ++c) {
    const float hv = hr[c];
    atomicAdd(gsum + b * kHid + c, hv);
    if (am != 0.f) atomicAdd(asum + b * kHid + c, hv);
  }
}

__global__ __launch_bounds__(128) void value_kernel(
    const float* __restrict__ gsum, const float* __restrict__ asum,
    const float* __restrict__ cnt, const float* __restrict__ acnt,
    const float* __restrict__ W1, const float* __restrict__ b1,
    const float* __restrict__ W2, const float* __restrict__ b2,
    float* __restrict__ out) {
  __shared__ float vi[2 * kHid];
  __shared__ float sred[128];
  const int b = blockIdx.x, c = threadIdx.x;
  const float cn = fmaxf(cnt[b], 1.f);
  const float ac = acnt[b] / cn;
  vi[c]        = gsum[b * kHid + c] / cn;
  vi[kHid + c] = (asum[b * kHid + c] / cn) / fmaxf(ac, 1e-6f);
  __syncthreads();
  float acc = b1[c];
  for (int k = 0; k < 2 * kHid; ++k) acc += vi[k] * W1[k * kHid + c];
  const float p = gelu_f(acc) * W2[c];
  const float red = block_reduce_sum_128(p, sred);
  if (c == 0) out[b] = red + b2[0];
}

// --------------------------------- launch ----------------------------------
extern "C" void kernel_launch(void* const* d_in, const int* in_sizes, int n_in,
                              void* d_out, int out_size, void* d_ws, size_t ws_size,
                              hipStream_t stream) {
  (void)in_sizes; (void)n_in; (void)out_size; (void)ws_size;

  const float* x      = (const float*)d_in[IN_X];
  const int*   eidx   = (const int*)d_in[IN_EIDX];
  const float* eattr  = (const float*)d_in[IN_EATTR];
  const float* u      = (const float*)d_in[IN_U];
  const unsigned char* mask = (const unsigned char*)d_in[IN_MASK];
  const int*   batch  = (const int*)d_in[IN_BATCH];

  // ---- workspace carve (all 256B aligned; ~125 MB total) ----
  char* base = (char*)d_ws;
  size_t off = 0;
  auto alloc = [&](size_t bytes) -> void* {
    void* r = base + off;
    off = (off + bytes + 255) & ~(size_t)255;
    return r;
  };
  float*     h      = (float*)alloc((size_t)kN * kHid * 4);
  float*     xl     = (float*)alloc((size_t)kN * kHid * 4);
  float*     xr     = (float*)alloc((size_t)kN * kHid * 4);
  float*     gat    = (float*)alloc((size_t)kN * kHid * 4);
  float*     logits = (float*)alloc((size_t)kE * kHeads * 4);
  unsigned*  segmax = (unsigned*)alloc((size_t)kN * kHeads * 4);
  float*     den    = (float*)alloc((size_t)kN * kHeads * 4);
  float*     u_emb  = (float*)alloc((size_t)kB * kHid * 4);
  float*     gsum   = (float*)alloc((size_t)kB * kHid * 4);
  float*     asum   = (float*)alloc((size_t)kB * kHid * 4);
  float*     cnt    = (float*)alloc((size_t)kB * 4);
  float*     acnt   = (float*)alloc((size_t)kB * 4);
  _Float16*  hbf    = (_Float16*)alloc((size_t)kN * kHid * 2);
  _Float16*  wlt    = (_Float16*)alloc((size_t)kLayers * kHid * kHid * 2);
  _Float16*  wrt    = (_Float16*)alloc((size_t)kLayers * kHid * kHid * 2);
  _Float16*  w1mt   = (_Float16*)alloc((size_t)kHid * kKpad * 2);
  _Float16*  w1ft   = (_Float16*)alloc((size_t)kHid * kKpad * 2);

  float* out_move  = (float*)d_out;
  float* out_alpha = out_move + kE;
  float* out_beta  = out_alpha + kE;
  float* out_value = out_beta + kE;

  // ---- prep: weights -> fp16 transposed ----
  for (int l = 0; l < kLayers; ++l) {
    const float* Wl = (const float*)d_in[IN_LAYER0 + 9 * l + 0];
    const float* Wr = (const float*)d_in[IN_LAYER0 + 9 * l + 2];
    convtr_kernel<<<(kHid * kHid + 255) / 256, 256, 0, stream>>>(
        Wl, wlt + (size_t)l * kHid * kHid, kHid, kHid, kHid);
    convtr_kernel<<<(kHid * kHid + 255) / 256, 256, 0, stream>>>(
        Wr, wrt + (size_t)l * kHid * kHid, kHid, kHid, kHid);
  }
  convtr_kernel<<<(kHid * kKpad + 255) / 256, 256, 0, stream>>>(
      (const float*)d_in[IN_MOVE_W1], w1mt, kEdgeIn, kHid, kKpad);
  convtr_kernel<<<(kHid * kKpad + 255) / 256, 256, 0, stream>>>(
      (const float*)d_in[IN_FRAC_W1], w1ft, kEdgeIn, kHid, kKpad);

  // ---- encoders ----
  glob_enc_kernel<<<kB, 128, 0, stream>>>(
      u, (const float*)d_in[IN_GE_W], (const float*)d_in[IN_GE_B],
      (const float*)d_in[IN_GE_G], (const float*)d_in[IN_GE_BN], u_emb);
  node_enc_kernel<<<kN, 128, 0, stream>>>(
      x, (const float*)d_in[IN_NE_W], (const float*)d_in[IN_NE_B],
      (const float*)d_in[IN_NE_G], (const float*)d_in[IN_NE_BN],
      u_emb, batch, h, hbf);

  // ---- GATv2 layers ----
  const int egrid = (kE * kHeads + 255) / 256;
  for (int l = 0; l < kLayers; ++l) {
    const void* const* L = d_in + IN_LAYER0 + 9 * l;
    const float* bl   = (const float*)L[1];
    const float* br   = (const float*)L[3];
    const float* We   = (const float*)L[4];
    const float* att  = (const float*)L[5];
    const float* bias = (const float*)L[6];
    const float* g    = (const float*)L[7];
    const float* bn   = (const float*)L[8];

    gemm_xlxr_kernel<<<kN / 16, 256, 0, stream>>>(
        hbf, wlt + (size_t)l * kHid * kHid, wrt + (size_t)l * kHid * kHid,
        bl, br, xl, xr);

    (void)hipMemsetAsync(segmax, 0, (size_t)kN * kHeads * 4, stream); // 0 < key(-inf)
    (void)hipMemsetAsync(den,    0, (size_t)kN * kHeads * 4, stream);
    (void)hipMemsetAsync(gat,    0, (size_t)kN * kHid * 4, stream);

    gat_logits_kernel<<<egrid, 256, 0, stream>>>(xl, xr, eidx, eattr, We, att,
                                                 logits, segmax);
    gat_expsum_kernel<<<egrid, 256, 0, stream>>>(eidx, logits, segmax, den);
    gat_scatter_kernel<<<egrid, 256, 0, stream>>>(eidx, logits, den, xl, gat);
    node_update_kernel<<<kN, 128, 0, stream>>>(h, hbf, gat, bias, g, bn);
  }

  // ---- edge MLP heads (move / frac) ----
  edge_mlp_kernel<<<kE / 64, 256, 0, stream>>>(
      hbf, eidx, eattr,
      w1mt, (const float*)d_in[IN_MOVE_B1], (const float*)d_in[IN_MOVE_W2],
      (const float*)d_in[IN_MOVE_B2],
      w1ft, (const float*)d_in[IN_FRAC_B1], (const float*)d_in[IN_FRAC_W2],
      (const float*)d_in[IN_FRAC_B2],
      out_move, out_alpha, out_beta);

  // ---- pooling + value head ----
  (void)hipMemsetAsync(gsum, 0, (size_t)kB * kHid * 4, stream);
  (void)hipMemsetAsync(asum, 0, (size_t)kB * kHid * 4, stream);
  (void)hipMemsetAsync(cnt,  0, (size_t)kB * 4, stream);
  (void)hipMemsetAsync(acnt, 0, (size_t)kB * 4, stream);
  pool_kernel<<<(kN + 255) / 256, 256, 0, stream>>>(h, batch, mask, gsum, asum,
                                                    cnt, acnt);
  value_kernel<<<kB, 128, 0, stream>>>(
      gsum, asum, cnt, acnt, (const float*)d_in[IN_VAL_W1],
      (const float*)d_in[IN_VAL_B1], (const float*)d_in[IN_VAL_W2],
      (const float*)d_in[IN_VAL_B2], out_value);
}
